// CrossTypeHGNN_40149354283050
// MI455X (gfx1250) — compile-verified, matrix operand order checked by reading, not run on hardware
//
#include <hip/hip_runtime.h>

#define N_NODES 6144
#define FDIM 64
#define BM 64
#define BK 32
#define LDA 40            // padded LDS row stride (bf16 elems)
#define LDU 72            // padded LDS row stride for u staging
#define KT_PER (N_NODES / BK)     // 192 k-steps per H matrix
#define KT_TOTAL (2 * KT_PER)     // 384 (two H matrices, concatenated K)
#define PF_DIST 8                 // prefetch distance in k-steps

typedef __attribute__((ext_vector_type(16))) __bf16 v16bf;
typedef __attribute__((ext_vector_type(8)))  float  v8f;
typedef __attribute__((ext_vector_type(4)))  float  f4;
typedef __attribute__((ext_vector_type(4)))  unsigned int u32x4;

#if defined(__HIP_DEVICE_COMPILE__) && defined(__has_builtin)
#if __has_builtin(__builtin_amdgcn_global_load_async_to_lds_b128)
#define USE_ASYNC_LDS 1
#endif
#endif

#ifdef USE_ASYNC_LDS
typedef int v4i __attribute__((vector_size(16)));
typedef __attribute__((address_space(1))) v4i as1_v4i;
typedef __attribute__((address_space(3))) v4i as3_v4i;
#endif

union V16U { u32x4 q[2]; v16bf v; };

__device__ __forceinline__ v16bf ldAB(const __bf16* p0, const __bf16* p1) {
  V16U u;
  u.q[0] = *(const u32x4*)p0;   // vec elems 0..7  -> VGPR 0..3 (K pairs)
  u.q[1] = *(const u32x4*)p1;   // vec elems 8..15 -> VGPR 4..7
  return u.v;
}

__device__ __forceinline__ v8f splat8(float v) {
  v8f r;
#pragma unroll
  for (int i = 0; i < 8; ++i) r[i] = v;
  return r;
}

__device__ __forceinline__ void cvt_store16(__bf16* dst, f4 a, f4 b) {
  union { __bf16 h[8]; u32x4 q; } pk;
#pragma unroll
  for (int i = 0; i < 4; ++i) pk.h[i] = (__bf16)a[i];
#pragma unroll
  for (int i = 0; i < 4; ++i) pk.h[4 + i] = (__bf16)b[i];
  *(u32x4*)dst = pk.q;
}

__device__ __forceinline__ void wait_async0() {
#ifdef USE_ASYNC_LDS
#if __has_builtin(__builtin_amdgcn_s_wait_asynccnt)
  __builtin_amdgcn_s_wait_asynccnt(0);
#else
  asm volatile("s_wait_asynccnt 0" ::: "memory");
#endif
#endif
}

struct TypeArgs {
  const float*  Ha;    // [N,N] fp32
  const float*  Hb;    // [N,N] fp32
  const __bf16* Ba;    // [64,N] bf16 transposed source a
  const __bf16* Bb;    // [64,N] bf16 transposed source b
  const __bf16* W;     // [64,64] bf16 row-major
  const float*  bias;  // [64] fp32
  __bf16*       outT;  // MODE 0: [64,N] bf16
  float*        outF;  // MODE 1: [N,64] fp32
};
struct LayerArgs { TypeArgs t[3]; };

struct PrepArgs {
  const float* x[3];
  __bf16*      xT[3];
  const float* W[6];
  __bf16*      Wb[6];
};

__global__ void hgnn_prep(PrepArgs p) {
  int idx = blockIdx.x * blockDim.x + threadIdx.x;
  int y = blockIdx.y;
  if (y < 3) {
    if (idx < N_NODES * FDIM) {
      int n = idx >> 6, f = idx & 63;
      p.xT[y][(size_t)f * N_NODES + n] = (__bf16)p.x[y][idx];
    }
  } else {
    if (idx < 6 * FDIM * FDIM) {
      int m = idx >> 12, e = idx & 4095;
      p.Wb[m][e] = (__bf16)p.W[m][e];
    }
  }
}

template <int MODE>
__global__ __launch_bounds__(256)
void hgnn_layer(LayerArgs args) {
  const TypeArgs ta = args.t[blockIdx.y];

  __shared__ __align__(16) __bf16 As[2][BM * LDA];
#ifdef USE_ASYNC_LDS
  __shared__ __align__(16) __bf16 Bs[2][FDIM * LDA];
#endif
  __shared__ __align__(16) __bf16 Us[BM * LDU];

  const int tid  = threadIdx.x;
  const int wave = tid >> 5;
  const int lane = tid & 31;
  const int lrow = lane & 15;
  const int hi   = lane >> 4;
  const int c0   = hi * 8;              // per-ISA 16-bit operand K chunk base
  const int rowg = (wave & 3) * 16;     // wave row group within 64
  const int colg = (wave >> 2) * 32;    // wave col group (0 or 32)
  const int blockRow = blockIdx.x * BM;

  // staging: 64x32 fp32 tile, 256 threads, 8 floats each / one 16B bf16 chunk each
  const int srow = tid >> 2;            // 0..63 (also B row n)
  const int sq   = tid & 3;             // 16B chunk within 32-elem row

  auto a_src = [&](int j) -> const float* {
    const float* Hm = (j < KT_PER) ? ta.Ha : ta.Hb;
    int kk = (j < KT_PER) ? j : j - KT_PER;
    return Hm + (size_t)(blockRow + srow) * N_NODES + kk * BK + sq * 8;
  };
  auto b_stage_src = [&](int j) -> const __bf16* {
    const __bf16* Bm = (j < KT_PER) ? ta.Ba : ta.Bb;
    int kk = (j < KT_PER) ? j : j - KT_PER;
    return Bm + (size_t)srow * N_NODES + kk * BK + sq * 8;
  };
  auto b_rd = [&](int j, int n) -> const __bf16* {
    const __bf16* Bm = (j < KT_PER) ? ta.Ba : ta.Bb;
    int kk = (j < KT_PER) ? j : j - KT_PER;
    return Bm + (size_t)n * N_NODES + kk * BK;
  };

#ifdef USE_ASYNC_LDS
  auto b_async = [&](int j, int buf) {
    // param1: global (AS1) int4*; param2: LDS (AS3) int4*.
    // Go through uintptr_t: generic VA == AS1 VA; low 32 bits of a flat LDS
    // address are the LDS offset (ISA 10.2 aperture rule), AS3 ptrs are 32-bit.
    as1_v4i* g = (as1_v4i*)(unsigned long long)(uintptr_t)b_stage_src(j);
    as3_v4i* l = (as3_v4i*)(unsigned int)(uintptr_t)&Bs[buf][srow * LDA + sq * 8];
    __builtin_amdgcn_global_load_async_to_lds_b128(g, l, 0, 0);
  };
#endif

  v8f acc[2];
  acc[0] = splat8(0.0f);
  acc[1] = splat8(0.0f);

  // ---- prologue: stage tile 0 ----
  {
    const f4* s = (const f4*)a_src(0);
    f4 x0 = __builtin_nontemporal_load(s);
    f4 x1 = __builtin_nontemporal_load(s + 1);
    cvt_store16(&As[0][srow * LDA + sq * 8], x0, x1);
#ifdef USE_ASYNC_LDS
    b_async(0, 0);
#endif
  }

  for (int j = 0; j < KT_TOTAL; ++j) {
    const int buf = j & 1;
    const bool hasNext = (j + 1) < KT_TOTAL;

    // issue next H tile loads early (overlaps with this iteration's WMMAs)
    f4 n0, n1;
    if (hasNext) {
      const f4* s = (const f4*)a_src(j + 1);
      n0 = __builtin_nontemporal_load(s);
      n1 = __builtin_nontemporal_load(s + 1);
    }
    // deep prefetch to cover HBM latency
    int jp = j + PF_DIST; if (jp >= KT_TOTAL) jp = KT_TOTAL - 1;
    __builtin_prefetch(a_src(jp), 0, 0);

    wait_async0();     // our B copy into Bs[buf] (issued last iter) is done
    __syncthreads();   // everyone's stores/copies into buffers[buf] visible

#ifdef USE_ASYNC_LDS
    // safe to refill buf^1 only after the barrier (readers of buf^1 finished)
    if (hasNext) b_async(j + 1, buf ^ 1);
#endif

    // ---- consume buffers[buf] ----
    const __bf16* arow = &As[buf][(rowg + lrow) * LDA];
    v16bf a = ldAB(arow + c0, arow + 16 + c0);
    v16bf b0, b1;
#ifdef USE_ASYNC_LDS
    {
      const __bf16* r0 = &Bs[buf][(colg + lrow) * LDA];
      const __bf16* r1 = &Bs[buf][(colg + 16 + lrow) * LDA];
      b0 = ldAB(r0 + c0, r0 + 16 + c0);
      b1 = ldAB(r1 + c0, r1 + 16 + c0);
    }
#else
    {
      const __bf16* r0 = b_rd(j, colg + lrow);
      const __bf16* r1 = b_rd(j, colg + 16 + lrow);
      b0 = ldAB(r0 + c0, r0 + 16 + c0);
      b1 = ldAB(r1 + c0, r1 + 16 + c0);
    }
#endif
    acc[0] = __builtin_amdgcn_wmma_f32_16x16x32_bf16(false, a, false, b0, (short)0, acc[0], false, false);
    acc[1] = __builtin_amdgcn_wmma_f32_16x16x32_bf16(false, a, false, b1, (short)0, acc[1], false, false);

    // stage next A tile into buf^1 (written only after this iteration's barrier)
    if (hasNext) cvt_store16(&As[buf ^ 1][srow * LDA + sq * 8], n0, n1);
  }

  // ---- fused per-type linear: out = u @ W^T + bias ----
  // restage u (this wave's 16x32 chunk) into full 64x64 bf16 tile in LDS
#pragma unroll
  for (int cb = 0; cb < 2; ++cb) {
#pragma unroll
    for (int r = 0; r < 8; ++r) {
      int m = rowg + r + hi * 8;
      int n = colg + cb * 16 + lrow;
      Us[m * LDU + n] = (__bf16)acc[cb][r];
    }
  }
  __syncthreads();

  v8f acc2[2];
#pragma unroll
  for (int cb = 0; cb < 2; ++cb) acc2[cb] = splat8(ta.bias[colg + cb * 16 + lrow]);

#pragma unroll
  for (int kh = 0; kh < 2; ++kh) {
    const __bf16* ar = &Us[(rowg + lrow) * LDU + kh * 32];
    v16bf a2 = ldAB(ar + c0, ar + 16 + c0);
#pragma unroll
    for (int cb = 0; cb < 2; ++cb) {
      const __bf16* wr = ta.W + (colg + cb * 16 + lrow) * FDIM + kh * 32;
      v16bf b2 = ldAB(wr + c0, wr + 16 + c0);
      acc2[cb] = __builtin_amdgcn_wmma_f32_16x16x32_bf16(false, a2, false, b2, (short)0, acc2[cb], false, false);
    }
  }

  // ---- writeback: C layout (m = r + 8*laneHi, n = lane&15) ----
#pragma unroll
  for (int cb = 0; cb < 2; ++cb) {
#pragma unroll
    for (int r = 0; r < 8; ++r) {
      int mg = blockRow + rowg + r + hi * 8;
      int n  = colg + cb * 16 + lrow;
      if (MODE == 0) {
        ta.outT[(size_t)n * N_NODES + mg] = (__bf16)acc2[cb][r];
      } else {
        ta.outF[(size_t)mg * FDIM + n] = acc2[cb][r];
      }
    }
  }
}

extern "C" void kernel_launch(void* const* d_in, const int* in_sizes, int n_in,
                              void* d_out, int out_size, void* d_ws, size_t ws_size,
                              hipStream_t stream) {
  (void)in_sizes; (void)n_in; (void)out_size; (void)ws_size;

  const float* x[3]  = {(const float*)d_in[0], (const float*)d_in[1], (const float*)d_in[2]};
  const float* H[6]  = {(const float*)d_in[3], (const float*)d_in[4], (const float*)d_in[5],
                        (const float*)d_in[6], (const float*)d_in[7], (const float*)d_in[8]};
  const float* W1[3] = {(const float*)d_in[9],  (const float*)d_in[10], (const float*)d_in[11]};
  const float* b1[3] = {(const float*)d_in[12], (const float*)d_in[13], (const float*)d_in[14]};
  const float* W2[3] = {(const float*)d_in[15], (const float*)d_in[16], (const float*)d_in[17]};
  const float* b2[3] = {(const float*)d_in[18], (const float*)d_in[19], (const float*)d_in[20]};

  char* ws = (char*)d_ws;
  const size_t XT_BYTES = (size_t)FDIM * N_NODES * 2;  // 786432
  __bf16* xT[3];
  __bf16* hT[3];
  for (int i = 0; i < 3; ++i) xT[i] = (__bf16*)(ws + (size_t)i * XT_BYTES);
  for (int i = 0; i < 3; ++i) hT[i] = (__bf16*)(ws + (3 + i) * XT_BYTES);
  __bf16* Wb[6];
  for (int i = 0; i < 6; ++i) Wb[i] = (__bf16*)(ws + 6 * XT_BYTES + (size_t)i * FDIM * FDIM * 2);

  PrepArgs pa;
  for (int i = 0; i < 3; ++i) { pa.x[i] = x[i]; pa.xT[i] = xT[i]; }
  for (int i = 0; i < 3; ++i) { pa.W[i] = W1[i]; pa.Wb[i] = Wb[i]; pa.W[3 + i] = W2[i]; pa.Wb[3 + i] = Wb[3 + i]; }
  hgnn_prep<<<dim3((N_NODES * FDIM + 255) / 256, 4), 256, 0, stream>>>(pa);

  LayerArgs l1;
  l1.t[0] = {H[0], H[1], xT[1], xT[2], Wb[0], b1[0], hT[0], nullptr};
  l1.t[1] = {H[2], H[3], xT[0], xT[2], Wb[1], b1[1], hT[1], nullptr};
  l1.t[2] = {H[4], H[5], xT[0], xT[1], Wb[2], b1[2], hT[2], nullptr};
  hgnn_layer<0><<<dim3(N_NODES / BM, 3), 256, 0, stream>>>(l1);

  float* out = (float*)d_out;
  LayerArgs l2;
  l2.t[0] = {H[0], H[1], hT[1], hT[2], Wb[3], b2[0], nullptr, out + 0 * (size_t)N_NODES * FDIM};
  l2.t[1] = {H[2], H[3], hT[0], hT[2], Wb[4], b2[1], nullptr, out + 1 * (size_t)N_NODES * FDIM};
  l2.t[2] = {H[4], H[5], hT[0], hT[1], Wb[5], b2[2], nullptr, out + 2 * (size_t)N_NODES * FDIM};
  hgnn_layer<1><<<dim3(N_NODES / BM, 3), 256, 0, stream>>>(l2);
}
